// Attention_48945447306134
// MI455X (gfx1250) — matrix operand order repaired and checked by reading, compiled, not run on hardware
//
#include <hip/hip_runtime.h>
#include <stdint.h>

// ---------------------------------------------------------------------------
// Types
// ---------------------------------------------------------------------------
typedef __bf16 bhalf;
typedef __attribute__((ext_vector_type(16))) __bf16 bhalf16;
typedef __attribute__((ext_vector_type(8)))  __bf16 bhalf8;
typedef __attribute__((ext_vector_type(4)))  __bf16 bhalf4;
typedef __attribute__((ext_vector_type(8)))  float  floatx8;
typedef __attribute__((ext_vector_type(4)))  unsigned int u32x4;
typedef __attribute__((ext_vector_type(8)))  int  i32x8;
typedef __attribute__((ext_vector_type(4)))  int  i32x4;

// Problem constants (fixed by the reference)
#define BATCH    4
#define SEQLEN   2048
#define EMBED    2048
#define NHEADS   16
#define NKV      4
#define HD       128
#define TOK      (BATCH * SEQLEN)          // 8192
#define KVDIM    (NKV * HD)                // 512

// ---------------------------------------------------------------------------
// CDNA5 data movers
// ---------------------------------------------------------------------------
// Per-lane async copy: 16B global -> 16B LDS, tracked by ASYNCcnt.
__device__ __forceinline__ void async_b128(void* lds_dst, const void* gsrc) {
  asm volatile("global_load_async_to_lds_b128 %0, %1, off"
               :: "v"((unsigned)(uintptr_t)lds_dst), "v"(gsrc)
               : "memory");
}
__device__ __forceinline__ void wait_async0() {
  asm volatile("s_wait_asynccnt 0x0" ::: "memory");
}

// Tensor Data Mover: DMA a 2D tile (bf16 elements) from global to LDS.
// D# group0 = count/lds/global/type, group1 = sizes/strides; groups 2/3
// zero (tile_dim2..4 = 0 -> unused).  One wave issues; EXEC ignored.
// 6-arg builtin form: (u32x4 g0, i32x8 g1, i32x4, i32x4, i32x8, i32 cpol)
__device__ __forceinline__ void tdm_load_2d_bf16(unsigned lds_addr, const void* gaddr,
                                                 unsigned tensor_d0, unsigned tensor_d1,
                                                 unsigned tile_d0, unsigned tile_d1,
                                                 unsigned stride0_elems) {
  const unsigned long long ga = (unsigned long long)(uintptr_t)gaddr;
  u32x4 g0;
  g0[0] = 1u;                                   // count=1, user descriptor
  g0[1] = lds_addr;                             // LDS byte address
  g0[2] = (unsigned)(ga & 0xffffffffu);         // global addr [31:0]
  g0[3] = (unsigned)((ga >> 32) & 0x01ffffffu)  // global addr [56:32]
          | (2u << 30);                         // type = 2 ("image")
  i32x8 g1;
  g1[0] = (int)(1u << 16);                      // data_size = 1 -> 2 bytes
  g1[1] = (int)((tensor_d0 & 0xffffu) << 16);   // tensor_dim0[15:0]
  g1[2] = (int)(((tensor_d0 >> 16) & 0xffffu) | ((tensor_d1 & 0xffffu) << 16));
  g1[3] = (int)(((tensor_d1 >> 16) & 0xffffu) | ((tile_d0 & 0xffffu) << 16));
  g1[4] = (int)(tile_d1 & 0xffffu);             // tile_dim1; tile_dim2 = 0
  g1[5] = (int)stride0_elems;                   // tensor_dim0_stride[31:0]
  g1[6] = 0;                                    // stride0[47:32], d1_stride lo
  g1[7] = 0;
  i32x4 z4; z4[0] = 0; z4[1] = 0; z4[2] = 0; z4[3] = 0;
  i32x8 z8;
#pragma unroll
  for (int i = 0; i < 8; i++) z8[i] = 0;
  __builtin_amdgcn_tensor_load_to_lds(g0, g1, z4, z4, z8, 0);
}
__device__ __forceinline__ void wait_tensor0() {
  __builtin_amdgcn_s_wait_tensorcnt(0);
}

// ---------------------------------------------------------------------------
// WMMA fragment helpers (CDNA5 wave32, 16x16x32 bf16; layouts per ISA 7.12.2)
// ---------------------------------------------------------------------------
__device__ __forceinline__ bhalf16 load_frag_a(const bhalf* p, int ld, int lane) {
  const int r = lane & 15, hi = lane >> 4;
  const bhalf* row = p + (size_t)r * ld;
  bhalf8 c0 = *(const bhalf8*)(row + hi * 8);
  bhalf8 c1 = *(const bhalf8*)(row + 16 + hi * 8);
  return __builtin_shufflevector(c0, c1, 0,1,2,3,4,5,6,7,8,9,10,11,12,13,14,15);
}

__device__ __forceinline__ bhalf16 load_frag_b(const bhalf* p, int ld, int lane) {
  const int n = lane & 15, hi = lane >> 4;
  const bhalf* row = p + (size_t)n * ld;
  bhalf8 c0 = *(const bhalf8*)(row + hi * 8);
  bhalf8 c1 = *(const bhalf8*)(row + 16 + hi * 8);
  return __builtin_shufflevector(c0, c1, 0,1,2,3,4,5,6,7,8,9,10,11,12,13,14,15);
}

__device__ __forceinline__ floatx8 wmma_bf16(bhalf16 a, bhalf16 b, floatx8 c) {
  return __builtin_amdgcn_wmma_f32_16x16x32_bf16(false, a, false, b, (short)0, c,
                                                 false, false);
}

// ---------------------------------------------------------------------------
// fp32 -> bf16 (n divisible by 4)
// ---------------------------------------------------------------------------
__global__ __launch_bounds__(256) void cvt_f32_bf16(const float* __restrict__ in,
                                                    bhalf* __restrict__ out, int n) {
  int i = (blockIdx.x * 256 + threadIdx.x) * 4;
  if (i >= n) return;
  float4 v = *(const float4*)(in + i);
  bhalf4 o;
  o[0] = (bhalf)v.x; o[1] = (bhalf)v.y; o[2] = (bhalf)v.z; o[3] = (bhalf)v.w;
  *(bhalf4*)(out + i) = o;
}

// Transpose + convert: W[K][N] fp32 -> Wt[N][K] bf16 (coalesced writes)
__global__ __launch_bounds__(256) void wtrans_kernel(const float* __restrict__ W,
                                                     bhalf* __restrict__ Wt,
                                                     int K, int N) {
  size_t t = (size_t)blockIdx.x * 256 + threadIdx.x;
  if (t >= (size_t)K * N) return;
  int k = (int)(t % K);
  int n = (int)(t / K);
  Wt[t] = (bhalf)W[(size_t)k * N + n];
}

// ---------------------------------------------------------------------------
// Tiled WMMA GEMM: C[M,N] = A[M,K] * Bt[N][K]^T.  Block tile 128x64, K-step
// 64, 256 threads = 8 waves (4x2), each wave a 32x32 tile.  Double-buffered:
// stage k+1 is async-DMA'd to LDS (GLOBAL_LOAD_ASYNC_TO_LDS_B128) while
// stage k runs on the matrix engine; S_WAIT_ASYNCCNT 0 fences each stage.
// ---------------------------------------------------------------------------
#define GBM 128
#define GBN 64
#define GBK 64

template <bool OUTF32>
__global__ __launch_bounds__(256) void gemm_bf16_kernel(const bhalf* __restrict__ A,
                                                        const bhalf* __restrict__ Bt,
                                                        void* __restrict__ Cv,
                                                        int M, int N, int K) {
  __shared__ __align__(16) bhalf As[2][GBM * GBK];   // 2 x 16 KB
  __shared__ __align__(16) bhalf Bs[2][GBN * GBK];   // 2 x  8 KB

  const int tid  = threadIdx.x;
  const int lane = tid & 31;
  const int wid  = tid >> 5;
  const int wm   = wid >> 1;    // 0..3
  const int wn   = wid & 1;     // 0..1
  const int m0   = blockIdx.y * GBM;
  const int n0   = blockIdx.x * GBN;

  const int ar = tid >> 1, ah = tid & 1;   // A: 128 rows x 2 halves of 32 el
  const int br = tid >> 2, bq = tid & 3;   // B: 64 rows x 4 segs of 16 el

  // Per-thread staging of one 128x64 A tile + 64x64 B tile (6 x b128 async)
  auto stage = [&](int kb, int buf) {
    const bhalf* ag = A + (size_t)(m0 + ar) * K + kb + ah * 32;
    bhalf* ad = &As[buf][ar * GBK + ah * 32];
#pragma unroll
    for (int u = 0; u < 4; u++) async_b128(ad + u * 8, ag + u * 8);
    const bhalf* bg = Bt + (size_t)(n0 + br) * K + kb + bq * 16;
    bhalf* bd = &Bs[buf][br * GBK + bq * 16];
#pragma unroll
    for (int u = 0; u < 2; u++) async_b128(bd + u * 8, bg + u * 8);
  };

  floatx8 acc[2][2];
#pragma unroll
  for (int i = 0; i < 2; i++)
#pragma unroll
    for (int j = 0; j < 2; j++)
#pragma unroll
      for (int e = 0; e < 8; e++) acc[i][j][e] = 0.0f;

  const int nkt = K / GBK;
  stage(0, 0);
  wait_async0();
  __syncthreads();

  for (int kt = 0; kt < nkt; kt++) {
    const int cur = kt & 1;
    if (kt + 1 < nkt) stage((kt + 1) * GBK, 1 - cur);  // overlap DMA w/ WMMA

    bhalf16 af[2][2], bfr[2][2];
#pragma unroll
    for (int ks = 0; ks < 2; ks++) {
#pragma unroll
      for (int i = 0; i < 2; i++)
        af[i][ks] = load_frag_a(&As[cur][(wm * 32 + i * 16) * GBK + ks * 32], GBK, lane);
#pragma unroll
      for (int j = 0; j < 2; j++)
        bfr[j][ks] = load_frag_b(&Bs[cur][(wn * 32 + j * 16) * GBK + ks * 32], GBK, lane);
    }
#pragma unroll
    for (int ks = 0; ks < 2; ks++)
#pragma unroll
      for (int i = 0; i < 2; i++)
#pragma unroll
        for (int j = 0; j < 2; j++)
          acc[i][j] = wmma_bf16(af[i][ks], bfr[j][ks], acc[i][j]);

    wait_async0();     // next stage landed in LDS
    __syncthreads();   // everyone done reading cur + sees next stage
  }

  // Epilogue: C/D layout -> lane holds column (lane&15), rows hi*8 + r
  const int cn = lane & 15, hi = lane >> 4;
#pragma unroll
  for (int i = 0; i < 2; i++)
#pragma unroll
    for (int j = 0; j < 2; j++)
#pragma unroll
      for (int r = 0; r < 8; r++) {
        const int row = m0 + wm * 32 + i * 16 + hi * 8 + r;
        const int col = n0 + wn * 32 + j * 16 + cn;
        if (OUTF32)
          ((float*)Cv)[(size_t)row * N + col] = acc[i][j][r];
        else
          ((bhalf*)Cv)[(size_t)row * N + col] = (bhalf)acc[i][j][r];
      }
}

// ---------------------------------------------------------------------------
// RoPE + layout shuffle (interleaved pairs; 1/sqrt(HD) folded into Q).
//   Qr [B*H ][S][HD] ; Kr [B*KV][S][HD] ; Vt [B*KV][HD][S]
// ---------------------------------------------------------------------------
__global__ __launch_bounds__(256) void rope_prep_kernel(const bhalf* __restrict__ qp,
                                                        const bhalf* __restrict__ kp,
                                                        const bhalf* __restrict__ vp,
                                                        const float* __restrict__ fcos,
                                                        const float* __restrict__ fsin,
                                                        bhalf* __restrict__ Qr,
                                                        bhalf* __restrict__ Kr,
                                                        bhalf* __restrict__ Vt) {
  const size_t t = (size_t)blockIdx.x * 256 + threadIdx.x; // < 2^23
  const int d2 = (int)(t & 63);
  const int h  = (int)((t >> 6) & 15);
  const int s  = (int)((t >> 10) & 2047);
  const int b  = (int)(t >> 21);

  const float c  = fcos[s * 64 + d2];
  const float sn = fsin[s * 64 + d2];
  const float qscale = 0.08838834764831845f; // 1/sqrt(128)

  {
    const size_t qi = ((size_t)(b * SEQLEN + s)) * EMBED + h * HD + 2 * d2;
    const float x0 = (float)qp[qi], x1 = (float)qp[qi + 1];
    const size_t qo = (((size_t)(b * NHEADS + h)) * SEQLEN + s) * HD + 2 * d2;
    Qr[qo]     = (bhalf)((x0 * c - x1 * sn) * qscale);
    Qr[qo + 1] = (bhalf)((x0 * sn + x1 * c) * qscale);
  }
  if (h < NKV) {
    const size_t ki = ((size_t)(b * SEQLEN + s)) * KVDIM + h * HD + 2 * d2;
    const float k0 = (float)kp[ki], k1 = (float)kp[ki + 1];
    const size_t ko = (((size_t)(b * NKV + h)) * SEQLEN + s) * HD + 2 * d2;
    Kr[ko]     = (bhalf)(k0 * c - k1 * sn);
    Kr[ko + 1] = (bhalf)(k0 * sn + k1 * c);

    const float v0 = (float)vp[ki], v1 = (float)vp[ki + 1];
    const size_t vo = (((size_t)(b * NKV + h)) * HD + 2 * d2) * (size_t)SEQLEN + s;
    Vt[vo]          = (bhalf)v0;
    Vt[vo + SEQLEN] = (bhalf)v1;
  }
}

// ---------------------------------------------------------------------------
// Flash-style attention.  Grid (S/64, NHEADS, BATCH), 128 threads = 4 waves.
// Double-buffered: wave 0 issues TDM (tensor_load_to_lds) for the NEXT 32-key
// K/V tile right after the barrier that published the current one, so the
// Tensor Data Mover runs concurrently with the score/softmax/PV WMMA work;
// S_WAIT_TENSORCNT 0 + workgroup barrier fence each step.
// ---------------------------------------------------------------------------
#define NKT (SEQLEN / 32)

__global__ __launch_bounds__(128) void attn_kernel(const bhalf* __restrict__ Qr,
                                                   const bhalf* __restrict__ Kr,
                                                   const bhalf* __restrict__ Vt,
                                                   bhalf* __restrict__ Aout) {
  __shared__ __align__(16) bhalf Ks[2][32 * HD];   // [key][d]   2 x 8 KB
  __shared__ __align__(16) bhalf Vs[2][HD * 32];   // [d][key]   2 x 8 KB
  __shared__ __align__(16) bhalf Ps[4][16 * 32];   // per-wave P [row][key]

  const int tid  = threadIdx.x;
  const int lane = tid & 31;
  const int w    = tid >> 5;
  const int cn   = lane & 15;
  const int hi   = lane >> 4;

  const int qb  = blockIdx.x;        // 0..31
  const int h   = blockIdx.y;        // 0..15
  const int b   = blockIdx.z;        // 0..3
  const int kvh = h >> 2;            // NUM_REP = 4
  const int q0  = qb * 64 + w * 16;

  const bhalf* Qbase = Qr + (((size_t)(b * NHEADS + h)) * SEQLEN + q0) * HD;
  const bhalf* Kbase = Kr + ((size_t)(b * NKV + kvh)) * SEQLEN * HD;
  const bhalf* Vbase = Vt + ((size_t)(b * NKV + kvh)) * HD * SEQLEN;

  // TDM staging of one K (32x128) + one V^T (128x32) tile into buffer `buf`
  auto stage_kv = [&](int kb, int buf) {
    tdm_load_2d_bf16((unsigned)(uintptr_t)&Ks[buf][0],
                     Kbase + (size_t)kb * 32 * HD,
                     /*tensor_d0=*/HD, /*tensor_d1=*/SEQLEN,
                     /*tile_d0=*/HD, /*tile_d1=*/32, /*stride0=*/HD);
    tdm_load_2d_bf16((unsigned)(uintptr_t)&Vs[buf][0],
                     Vbase + (size_t)kb * 32,
                     /*tensor_d0=*/SEQLEN, /*tensor_d1=*/HD,
                     /*tile_d0=*/32, /*tile_d1=*/HD, /*stride0=*/SEQLEN);
  };

  bhalf16 qf[4];
#pragma unroll
  for (int kk = 0; kk < 4; kk++)
    qf[kk] = load_frag_a(Qbase + kk * 32, HD, lane);

  float mrow[8], lrow[8];
  floatx8 o[8];
#pragma unroll
  for (int i = 0; i < 8; i++) { mrow[i] = -3.0e38f; lrow[i] = 0.0f; }
#pragma unroll
  for (int dt = 0; dt < 8; dt++)
#pragma unroll
    for (int e = 0; e < 8; e++) o[dt][e] = 0.0f;

  if (w == 0) {
    stage_kv(0, 0);
    wait_tensor0();
  }
  __syncthreads();

  for (int kb = 0; kb < NKT; kb++) {
    const int cur = kb & 1;
    if (w == 0 && kb + 1 < NKT) stage_kv(kb + 1, 1 - cur);  // overlap TDM

    const bhalf* KsC = &Ks[cur][0];
    const bhalf* VsC = &Vs[cur][0];

    // Scores: S = Q * K^T, 32 keys (two 16-wide n-tiles), K-dim = 128.
    // All 8 B-fragments preloaded into distinct regs -> batched ds_loads.
    bhalf16 kf[2][4];
#pragma unroll
    for (int kk = 0; kk < 4; kk++) {
      kf[0][kk] = load_frag_b(KsC + 0 * 16 * HD + kk * 32, HD, lane);
      kf[1][kk] = load_frag_b(KsC + 1 * 16 * HD + kk * 32, HD, lane);
    }
    floatx8 s0, s1;
#pragma unroll
    for (int e = 0; e < 8; e++) { s0[e] = 0.0f; s1[e] = 0.0f; }
#pragma unroll
    for (int kk = 0; kk < 4; kk++) {
      s0 = wmma_bf16(qf[kk], kf[0][kk], s0);
      s1 = wmma_bf16(qf[kk], kf[1][kk], s1);
    }

    // Online softmax in the C-fragment layout (row = VGPR idx, half-wave
    // shfl_xor 1/2/4/8 reduces across the 16 lanes holding one row).
#pragma unroll
    for (int i = 0; i < 8; i++) {
      float mx = fmaxf(s0[i], s1[i]);
      mx = fmaxf(mx, __shfl_xor(mx, 1, 32));
      mx = fmaxf(mx, __shfl_xor(mx, 2, 32));
      mx = fmaxf(mx, __shfl_xor(mx, 4, 32));
      mx = fmaxf(mx, __shfl_xor(mx, 8, 32));
      const float mnew  = fmaxf(mrow[i], mx);
      const float alpha = __expf(mrow[i] - mnew);
      mrow[i] = mnew;
      const float p0 = __expf(s0[i] - mnew);
      const float p1 = __expf(s1[i] - mnew);
      float rs = p0 + p1;
      rs += __shfl_xor(rs, 1, 32);
      rs += __shfl_xor(rs, 2, 32);
      rs += __shfl_xor(rs, 4, 32);
      rs += __shfl_xor(rs, 8, 32);
      lrow[i] = lrow[i] * alpha + rs;
      Ps[w][(hi * 8 + i) * 32 + cn]      = (bhalf)p0;
      Ps[w][(hi * 8 + i) * 32 + 16 + cn] = (bhalf)p1;
#pragma unroll
      for (int dt = 0; dt < 8; dt++) o[dt][i] *= alpha;
    }

    // PV: O += P(16x32) * V(32x128); V B-fragments in groups of 4
    bhalf16 pa = load_frag_a(&Ps[w][0], 32, lane);
#pragma unroll
    for (int g = 0; g < 2; g++) {
      bhalf16 vb[4];
#pragma unroll
      for (int q = 0; q < 4; q++)
        vb[q] = load_frag_b(VsC + (g * 4 + q) * 16 * 32, 32, lane);
#pragma unroll
      for (int q = 0; q < 4; q++)
        o[g * 4 + q] = wmma_bf16(pa, vb[q], o[g * 4 + q]);
    }

    if (w == 0) wait_tensor0();  // next K/V tile fully in LDS
    __syncthreads();             // publish next tile; done reading cur
  }

  // Epilogue: normalize and write bf16 to Aout[B*S][NHEADS*HD]
#pragma unroll
  for (int i = 0; i < 8; i++) {
    const float inv = 1.0f / lrow[i];
    const int srow  = q0 + hi * 8 + i;
    const size_t base = ((size_t)b * SEQLEN + srow) * (size_t)EMBED + h * HD;
#pragma unroll
    for (int dt = 0; dt < 8; dt++)
      Aout[base + dt * 16 + cn] = (bhalf)(o[dt][i] * inv);
  }
}

// ---------------------------------------------------------------------------
// Host-side orchestration
// ---------------------------------------------------------------------------
extern "C" void kernel_launch(void* const* d_in, const int* in_sizes, int n_in,
                              void* d_out, int out_size, void* d_ws, size_t ws_size,
                              hipStream_t stream) {
  (void)in_sizes; (void)n_in; (void)out_size; (void)ws_size;

  const float* x    = (const float*)d_in[0];
  const float* fcos = (const float*)d_in[1];
  const float* fsin = (const float*)d_in[2];
  const float* wq   = (const float*)d_in[3];
  const float* wk   = (const float*)d_in[4];
  const float* wv   = (const float*)d_in[5];
  const float* wo   = (const float*)d_in[6];
  float* out = (float*)d_out;

  char* p = (char*)d_ws;
  auto take = [&](size_t elems) {
    bhalf* q = (bhalf*)p;
    p += (elems * sizeof(bhalf) + 255) & ~(size_t)255;
    return q;
  };
  bhalf* xb    = take((size_t)TOK * EMBED);
  bhalf* wqT   = take((size_t)EMBED * EMBED);
  bhalf* wkT   = take((size_t)KVDIM * EMBED);
  bhalf* wvT   = take((size_t)KVDIM * EMBED);
  bhalf* woT   = take((size_t)EMBED * EMBED);
  bhalf* qproj = take((size_t)TOK * EMBED);
  bhalf* kproj = take((size_t)TOK * KVDIM);
  bhalf* vproj = take((size_t)TOK * KVDIM);
  bhalf* Qr    = take((size_t)TOK * EMBED);
  bhalf* Kr    = take((size_t)BATCH * NKV * SEQLEN * HD);
  bhalf* Vt    = take((size_t)BATCH * NKV * HD * SEQLEN);
  bhalf* attn  = xb;  // reuse: xb is dead after the QKV GEMMs

  {
    const int n = TOK * EMBED;
    cvt_f32_bf16<<<n / 1024, 256, 0, stream>>>(x, xb, n);
  }
  wtrans_kernel<<<(EMBED * EMBED) / 256, 256, 0, stream>>>(wq, wqT, EMBED, EMBED);
  wtrans_kernel<<<(EMBED * KVDIM) / 256, 256, 0, stream>>>(wk, wkT, EMBED, KVDIM);
  wtrans_kernel<<<(EMBED * KVDIM) / 256, 256, 0, stream>>>(wv, wvT, EMBED, KVDIM);
  wtrans_kernel<<<(EMBED * EMBED) / 256, 256, 0, stream>>>(wo, woT, EMBED, EMBED);

  gemm_bf16_kernel<false><<<dim3(EMBED / GBN, TOK / GBM), 256, 0, stream>>>(
      xb, wqT, qproj, TOK, EMBED, EMBED);
  gemm_bf16_kernel<false><<<dim3(KVDIM / GBN, TOK / GBM), 256, 0, stream>>>(
      xb, wkT, kproj, TOK, KVDIM, EMBED);
  gemm_bf16_kernel<false><<<dim3(KVDIM / GBN, TOK / GBM), 256, 0, stream>>>(
      xb, wvT, vproj, TOK, KVDIM, EMBED);

  {
    const size_t total = (size_t)BATCH * SEQLEN * NHEADS * (HD / 2); // 2^23
    rope_prep_kernel<<<(unsigned)(total / 256), 256, 0, stream>>>(
        qproj, kproj, vproj, fcos, fsin, Qr, Kr, Vt);
  }

  attn_kernel<<<dim3(SEQLEN / 64, NHEADS, BATCH), 128, 0, stream>>>(Qr, Kr, Vt, attn);

  gemm_bf16_kernel<true><<<dim3(EMBED / GBN, TOK / GBM), 256, 0, stream>>>(
      attn, woT, out, TOK, EMBED, EMBED);
}
